// FKAConvNetwork_21345987461171
// MI455X (gfx1250) — matrix operand order repaired and checked by reading, not compiled, not run on hardware
//
#include <hip/hip_runtime.h>
#include <hip/hip_bf16.h>
#include <math.h>

typedef __attribute__((ext_vector_type(16))) _Float16 v16h;
typedef __attribute__((ext_vector_type(8)))  _Float16 v8h;
typedef __attribute__((ext_vector_type(4)))  _Float16 h4;
typedef __attribute__((ext_vector_type(8)))  float    v8f;

#define KS 16

// ---------------------------------------------------------------------------
// Farthest point sampling: one block per batch. pos [B,3,N] -> sup [B,3,nsup]
// ---------------------------------------------------------------------------
__global__ __launch_bounds__(256) void fps_kernel(const float* __restrict__ pos,
                                                  int N, int nsup,
                                                  float* __restrict__ sup) {
  int b = blockIdx.x;
  int tid = threadIdx.x;
  __shared__ float s_val[256];
  __shared__ int   s_id[256];
  __shared__ float s_last[3];
  const float* px = pos + (size_t)b * 3 * N;
  float* so = sup + (size_t)b * 3 * nsup;
  float dmin[32];
  int per = (N + 255) >> 8;
  for (int i = 0; i < 32; ++i) dmin[i] = 1e10f;
  if (tid == 0) {
    s_last[0] = px[0]; s_last[1] = px[N]; s_last[2] = px[2 * N];
    so[0] = px[0]; so[nsup] = px[N]; so[2 * nsup] = px[2 * N];
  }
  __syncthreads();
  for (int it = 1; it < nsup; ++it) {
    float lx = s_last[0], ly = s_last[1], lz = s_last[2];
    float best = -1.0f; int bid = 0x7fffffff;
    for (int i = 0; i < per; ++i) {
      int j = tid + (i << 8);
      if (j < N) {
        float dx = px[j] - lx, dy = px[N + j] - ly, dz = px[2 * N + j] - lz;
        float d = dx * dx + dy * dy + dz * dz;
        if (d < dmin[i]) dmin[i] = d;
        if (dmin[i] > best || (dmin[i] == best && j < bid)) { best = dmin[i]; bid = j; }
      }
    }
    s_val[tid] = best; s_id[tid] = bid;
    __syncthreads();
    for (int s = 128; s > 0; s >>= 1) {
      if (tid < s) {
        if (s_val[tid + s] > s_val[tid] ||
            (s_val[tid + s] == s_val[tid] && s_id[tid + s] < s_id[tid])) {
          s_val[tid] = s_val[tid + s]; s_id[tid] = s_id[tid + s];
        }
      }
      __syncthreads();
    }
    if (tid == 0) {
      int j = s_id[0];
      s_last[0] = px[j]; s_last[1] = px[N + j]; s_last[2] = px[2 * N + j];
      so[it] = px[j]; so[nsup + it] = px[N + j]; so[2 * nsup + it] = px[2 * N + j];
    }
    __syncthreads();
  }
}

// ---------------------------------------------------------------------------
// Brute-force KNN (top-16 smallest d2), LDS-tiled points.
// pts [B,3,Np], qs [B,3,Nq] -> idx [B,Nq,16]
// ---------------------------------------------------------------------------
__global__ __launch_bounds__(256) void knn_kernel(const float* __restrict__ pts, int Np,
                                                  const float* __restrict__ qs, int Nq,
                                                  int* __restrict__ outidx) {
  int b = blockIdx.y;
  int q = blockIdx.x * 256 + threadIdx.x;
  const float* pp = pts + (size_t)b * 3 * Np;
  const float* qq = qs + (size_t)b * 3 * Nq;
  bool act = (q < Nq);
  float qx = 0.f, qy = 0.f, qz = 0.f;
  if (act) { qx = qq[q]; qy = qq[Nq + q]; qz = qq[2 * Nq + q]; }
  float bd[16]; int bi[16];
  for (int i = 0; i < 16; ++i) { bd[i] = 1e30f; bi[i] = 0; }
  __shared__ float sx[256], sy[256], sz[256];
  for (int base = 0; base < Np; base += 256) {
    int j = base + threadIdx.x;
    if (j < Np) { sx[threadIdx.x] = pp[j]; sy[threadIdx.x] = pp[Np + j]; sz[threadIdx.x] = pp[2 * Np + j]; }
    __syncthreads();
    int cnt = (Np - base < 256) ? (Np - base) : 256;
    if (act) {
      for (int t = 0; t < cnt; ++t) {
        float dx = qx - sx[t], dy = qy - sy[t], dz = qz - sz[t];
        float d = dx * dx + dy * dy + dz * dz;
        if (d < bd[15]) {
          int p = 15;
          while (p > 0 && bd[p - 1] > d) { bd[p] = bd[p - 1]; bi[p] = bi[p - 1]; --p; }
          bd[p] = d; bi[p] = base + t;
        }
      }
    }
    __syncthreads();
  }
  if (act) {
    for (int i = 0; i < 16; ++i) outidx[((size_t)b * Nq + q) * 16 + i] = bi[i];
  }
}

// ---------------------------------------------------------------------------
// FKAConv stage 1: local coords, dw (sigmoid weighting), fc1, IN stats.
// grid (Nq/16, B), block 256 = 16 queries x 16 neighbors
// ---------------------------------------------------------------------------
__global__ __launch_bounds__(256) void fka_stage1(
    const float* __restrict__ pos, const float* __restrict__ sup,
    const int* __restrict__ idx,
    const float* __restrict__ fc1, const float* __restrict__ alpha,
    const float* __restrict__ beta, const float* __restrict__ nrad,
    int Npts, int Nq,
    float* __restrict__ dwbuf, float* __restrict__ mat1,
    float* __restrict__ osum, float* __restrict__ osq) {
  int b = blockIdx.y, tid = threadIdx.x;
  int q = tid >> 4, j = tid & 15;
  int n = blockIdx.x * 16 + q;
  __shared__ float s_fc[48];
  __shared__ float s_dw[16][16];
  __shared__ float s_sum[16], s_sq[16];
  if (tid < 48) s_fc[tid] = fc1[tid];
  if (tid < 16) { s_sum[tid] = 0.f; s_sq[tid] = 0.f; }
  int id = idx[((size_t)b * Nq + n) * 16 + j];
  float dx = pos[((size_t)b * 3 + 0) * Npts + id] - sup[((size_t)b * 3 + 0) * Nq + n];
  float dy = pos[((size_t)b * 3 + 1) * Npts + id] - sup[((size_t)b * 3 + 1) * Nq + n];
  float dz = pos[((size_t)b * 3 + 2) * Npts + id] - sup[((size_t)b * 3 + 2) * Nq + n];
  float dist = sqrtf(dx * dx + dy * dy + dz * dz);
  float z = -alpha[0] * dist + beta[0];
  float dwraw = 1.f / (1.f + expf(-z));
  s_dw[q][j] = dwraw;
  __syncthreads();
  float s = 0.f;
  for (int jj = 0; jj < 16; ++jj) s += s_dw[q][jj];
  s = s + ((s == 0.f) ? 1.f : 0.f) + 1e-6f;
  float dwv = dwraw / s * 16.f;
  dwbuf[((size_t)b * Nq + n) * 16 + j] = dwv;
  float inr = 1.f / nrad[0];
  dx *= inr; dy *= inr; dz *= inr;
#pragma unroll
  for (int k = 0; k < 16; ++k) {
    float m = s_fc[k * 3 + 0] * dx + s_fc[k * 3 + 1] * dy + s_fc[k * 3 + 2] * dz;
    mat1[(((size_t)b * 16 + k) * Nq + n) * 16 + j] = m;
    atomicAdd(&s_sum[k], m);
    atomicAdd(&s_sq[k], m * m);
  }
  __syncthreads();
  if (tid < 16) {
    atomicAdd(&osum[b * 16 + tid], s_sum[tid]);
    atomicAdd(&osq[b * 16 + tid], s_sq[tid]);
  }
}

// ---------------------------------------------------------------------------
// Instance-norm finalize: scale/shift per (b,k). <<<1,32>>>
// ---------------------------------------------------------------------------
__global__ void in_finalize(const float* __restrict__ ssum, const float* __restrict__ ssq,
                            const float* __restrict__ g, const float* __restrict__ bb,
                            float cntinv, float* __restrict__ ss) {
  int t = threadIdx.x;
  if (t < 32) {
    int k = t & 15;
    float mean = ssum[t] * cntinv;
    float var = ssq[t] * cntinv - mean * mean;
    var = fmaxf(var, 0.f);
    float sc = g[k] * rsqrtf(var + 1e-5f);
    ss[t * 2] = sc;
    ss[t * 2 + 1] = bb[k] - mean * sc;
  }
}

// ---------------------------------------------------------------------------
// FKAConv stage 3/5: normalize+relu, dw-weighted maxpool over neighbors,
// concat, 16x32 FC. mode 0: raw output + IN stats. mode 1: relu(out)*dw.
// ---------------------------------------------------------------------------
__global__ __launch_bounds__(256) void fka_stage_fused(
    const float* __restrict__ mat_in, const float* __restrict__ dw,
    const float* __restrict__ ss, const float* __restrict__ fcw,
    int Nq, int mode,
    float* __restrict__ mat_out, float* osum, float* osq) {
  int b = blockIdx.y, tid = threadIdx.x;
  int q = tid >> 4, j = tid & 15;
  int n = blockIdx.x * 16 + q;
  __shared__ float s_fc[16 * 32];
  __shared__ float s_md[16][16][16];  // [q][j][k]
  __shared__ float s_mp[16][16];
  __shared__ float s_sum[16], s_sq[16];
  for (int t = tid; t < 512; t += 256) s_fc[t] = fcw[t];
  if (tid < 16) { s_sum[tid] = 0.f; s_sq[tid] = 0.f; }
  float v[16];
  float dwv = dw[((size_t)b * Nq + n) * 16 + j];
#pragma unroll
  for (int k = 0; k < 16; ++k) {
    float sc = ss[(b * 16 + k) * 2];
    float sh = ss[(b * 16 + k) * 2 + 1];
    float r = mat_in[(((size_t)b * 16 + k) * Nq + n) * 16 + j] * sc + sh;
    r = (r > 0.f) ? r : 0.f;
    v[k] = r;
    s_md[q][j][k] = r * dwv;
  }
  __syncthreads();
  {
    float mp = -1e30f;
    for (int jj = 0; jj < 16; ++jj) mp = fmaxf(mp, s_md[q][jj][j]);
    s_mp[q][j] = mp;
  }
  __syncthreads();
#pragma unroll
  for (int k2 = 0; k2 < 16; ++k2) {
    float acc = 0.f;
#pragma unroll
    for (int c = 0; c < 16; ++c) acc += s_fc[k2 * 32 + c] * v[c];
#pragma unroll
    for (int c = 0; c < 16; ++c) acc += s_fc[k2 * 32 + 16 + c] * s_mp[q][c];
    if (mode == 0) {
      mat_out[(((size_t)b * 16 + k2) * Nq + n) * 16 + j] = acc;
      atomicAdd(&s_sum[k2], acc);
      atomicAdd(&s_sq[k2], acc * acc);
    } else {
      float o = (acc > 0.f) ? acc : 0.f;
      mat_out[(((size_t)b * 16 + k2) * Nq + n) * 16 + j] = o * dwv;
    }
  }
  __syncthreads();
  if (mode == 0 && tid < 16) {
    atomicAdd(&osum[b * 16 + tid], s_sum[tid]);
    atomicAdd(&osq[b * 16 + tid], s_sq[tid]);
  }
}

// ---------------------------------------------------------------------------
// fs einsum: fs[b, c*16+k, n] = sum_j x[b,c,idx[n,j]] * mat3[b,k,n,j]
// grid (Nq, B), block 256
// ---------------------------------------------------------------------------
__global__ __launch_bounds__(256) void fs_kernel(
    const float* __restrict__ x, int C, int Npts,
    const int* __restrict__ idx, const float* __restrict__ mat3,
    int Nq, float* __restrict__ fs) {
  int n = blockIdx.x, b = blockIdx.y;
  int tid = threadIdx.x;
  int r = tid >> 4, s = tid & 15;
  __shared__ float s_m3[16][16];  // [k][j]
  __shared__ int   s_id[16];
  __shared__ float s_xn[16][17];  // [c][j]
  s_m3[r][s] = mat3[(((size_t)b * 16 + r) * Nq + n) * 16 + s];
  if (tid < 16) s_id[tid] = idx[((size_t)b * Nq + n) * 16 + tid];
  __syncthreads();
  int CK = C * 16;
  for (int cc = 0; cc < C; cc += 16) {
    int c = cc + r;
    s_xn[r][s] = (c < C) ? x[((size_t)b * C + c) * Npts + s_id[s]] : 0.f;
    __syncthreads();
    float f = 0.f;
#pragma unroll
    for (int j = 0; j < 16; ++j) f += s_xn[r][j] * s_m3[s][j];
    if (c < C) fs[((size_t)b * CK + (size_t)c * 16 + s) * Nq + n] = f;
    __syncthreads();
  }
}

// ---------------------------------------------------------------------------
// WMMA GEMM:  out[M,N] = A[M,K] x Bm[K,N]  (+bias)(+BN)(+res)(+relu)
// f32 memory, f16 WMMA operands, f32 accumulation. wave32, 4 waves/block.
// Block tile 64(M) x 32(N), K-step 64: each wave holds 2 accumulators and
// issues 4 v_wmma per K iteration. LDS tiles padded to 72-half rows (144 B,
// 16B-aligned at every multiple-of-8-half offset) so fragments load as
// ds_load_b128 pairs; B is stored transposed [n][k] for contiguous K runs.
// Staging issues ALL global_load_b128s into registers before converting,
// so loadcnt waits are amortized instead of serializing every load.
// ---------------------------------------------------------------------------
__global__ __launch_bounds__(128) void gemm_wmma(
    const float* __restrict__ A, const float* __restrict__ Bm,
    const float* __restrict__ bias,
    const float* __restrict__ g, const float* __restrict__ bb,
    const float* __restrict__ rm, const float* __restrict__ rv,
    const float* __restrict__ res, float* __restrict__ out,
    int M, int N, int K, int relu) {
  int tid = threadIdx.x;
  int wave = tid >> 5;
  int lane = tid & 31;
  int m0 = blockIdx.y * 64;
  int n0 = blockIdx.x * 32;
  __shared__ __attribute__((aligned(16))) _Float16 As[64][72];
  __shared__ __attribute__((aligned(16))) _Float16 Bt[32][72];  // [n][k]
  v8f acc0 = {0.f, 0.f, 0.f, 0.f, 0.f, 0.f, 0.f, 0.f};
  v8f acc1 = {0.f, 0.f, 0.f, 0.f, 0.f, 0.f, 0.f, 0.f};
  // A-tile staging coordinates (fast path): 8 float4 per thread
  int ar = tid >> 4;             // row 0..7 step 8
  int ac4 = (tid & 15) << 2;     // col group
  // B-tile staging coordinates (fast path): 4 float4 per thread
  int br = tid >> 3;             // k-row 0..15 step 16
  int bc4 = (tid & 7) << 2;      // n group
  for (int k0 = 0; k0 < K; k0 += 64) {
    // ---- stage A tile 64x64 ----
    if (m0 + 64 <= M && k0 + 64 <= K) {
      float4 av[8];
#pragma unroll
      for (int i = 0; i < 8; ++i) {
        av[i] = *(const float4*)(A + (size_t)(m0 + ar + i * 8) * K + k0 + ac4);
      }
#pragma unroll
      for (int i = 0; i < 8; ++i) {
        h4 hv;
        hv[0] = (_Float16)av[i].x; hv[1] = (_Float16)av[i].y;
        hv[2] = (_Float16)av[i].z; hv[3] = (_Float16)av[i].w;
        *(h4*)&As[ar + i * 8][ac4] = hv;
      }
    } else {
      for (int t = tid; t < 4096; t += 128) {
        int r = t >> 6, c = t & 63;
        int m = m0 + r, kk = k0 + c;
        As[r][c] = (m < M && kk < K) ? (_Float16)A[(size_t)m * K + kk] : (_Float16)0.f;
      }
    }
    // ---- stage B tile 64x32, transposed into Bt[n][k] ----
    if (k0 + 64 <= K && n0 + 32 <= N) {
      float4 bv[4];
#pragma unroll
      for (int i = 0; i < 4; ++i) {
        bv[i] = *(const float4*)(Bm + (size_t)(k0 + br + i * 16) * N + n0 + bc4);
      }
#pragma unroll
      for (int i = 0; i < 4; ++i) {
        int r = br + i * 16;
        Bt[bc4 + 0][r] = (_Float16)bv[i].x;
        Bt[bc4 + 1][r] = (_Float16)bv[i].y;
        Bt[bc4 + 2][r] = (_Float16)bv[i].z;
        Bt[bc4 + 3][r] = (_Float16)bv[i].w;
      }
    } else {
      for (int t = tid; t < 2048; t += 128) {
        int r = t >> 5, c = t & 31;
        int kk = k0 + r, n = n0 + c;
        Bt[c][r] = (kk < K && n < N) ? (_Float16)Bm[(size_t)kk * N + n] : (_Float16)0.f;
      }
    }
    if (k0 + 64 < K) {
      __builtin_prefetch(&Bm[(size_t)(k0 + 64) * N + n0], 0, 1);
      __builtin_prefetch(&A[(size_t)(m0 + ar) * K + k0 + 64], 0, 1);
    }
    __syncthreads();
    // A fragment (16x32 f16): lane<16 holds M=lane, K={0..7,16..23};
    // lane>=16 holds M=lane-16, K={8..15,24..31}.  (+o for second K half)
    int mrow = (lane & 15) + (wave << 4);
    int kb = (lane < 16) ? 0 : 8;
    int ncol = lane & 15;
#pragma unroll
    for (int h = 0; h < 2; ++h) {
      int o = h << 5;
      v8h alo = *(const v8h*)&As[mrow][o + kb];
      v8h ahi = *(const v8h*)&As[mrow][o + kb + 16];
      v16h afrag = __builtin_shufflevector(alo, ahi, 0, 1, 2, 3, 4, 5, 6, 7,
                                           8, 9, 10, 11, 12, 13, 14, 15);
      // B fragment (32x16 f16): lane<16 -> N=lane, K=o+{0..7,16..23};
      // lane>=16 -> N=lane-16, K=o+{8..15,24..31}.
      int kbb = o + ((lane < 16) ? 0 : 8);
      v8h b0lo = *(const v8h*)&Bt[ncol][kbb];
      v8h b0hi = *(const v8h*)&Bt[ncol][kbb + 16];
      v16h bfrag0 = __builtin_shufflevector(b0lo, b0hi, 0, 1, 2, 3, 4, 5, 6, 7,
                                            8, 9, 10, 11, 12, 13, 14, 15);
      v8h b1lo = *(const v8h*)&Bt[16 + ncol][kbb];
      v8h b1hi = *(const v8h*)&Bt[16 + ncol][kbb + 16];
      v16h bfrag1 = __builtin_shufflevector(b1lo, b1hi, 0, 1, 2, 3, 4, 5, 6, 7,
                                            8, 9, 10, 11, 12, 13, 14, 15);
      acc0 = __builtin_amdgcn_wmma_f32_16x16x32_f16(false, afrag, false, bfrag0,
                                                    (short)0, acc0, false, false);
      acc1 = __builtin_amdgcn_wmma_f32_16x16x32_f16(false, afrag, false, bfrag1,
                                                    (short)0, acc1, false, false);
    }
    __syncthreads();
  }
  // D layout: VGPR r -> lanes 0-15: (M=r, N=lane); lanes 16-31: (M=8+r, N=lane-16)
#pragma unroll
  for (int r = 0; r < 8; ++r) {
    int m = m0 + (wave << 4) + r + ((lane < 16) ? 0 : 8);
    int na = n0 + (lane & 15);
    int nb = na + 16;
    if (m < M) {
      float inv = (g) ? rsqrtf(rv[m] + 1e-5f) : 0.f;
      if (na < N) {
        float v = acc0[r];
        if (bias) v += bias[m];
        if (g) v = (v - rm[m]) * inv * g[m] + bb[m];
        if (res) v += res[(size_t)m * N + na];
        if (relu) v = fmaxf(v, 0.f);
        out[(size_t)m * N + na] = v;
      }
      if (nb < N) {
        float v = acc1[r];
        if (bias) v += bias[m];
        if (g) v = (v - rm[m]) * inv * g[m] + bb[m];
        if (res) v += res[(size_t)m * N + nb];
        if (relu) v = fmaxf(v, 0.f);
        out[(size_t)m * N + nb] = v;
      }
    }
  }
}

// ---------------------------------------------------------------------------
// Shortcut spatial downsample: out[b,c,n] = max_j in[b,c,idx[n,j]]
// ---------------------------------------------------------------------------
__global__ void maxpool_kernel(const float* __restrict__ in, const int* __restrict__ idx,
                               int C, int Nin, int Nq, float* __restrict__ out) {
  int t = blockIdx.x * 256 + threadIdx.x;
  int total = 2 * C * Nq;
  if (t < total) {
    int n = t % Nq;
    int c = (t / Nq) % C;
    int b = t / (Nq * C);
    float mx = -1e30f;
    for (int j = 0; j < 16; ++j) {
      int id = idx[((size_t)b * Nq + n) * 16 + j];
      mx = fmaxf(mx, in[((size_t)b * C + c) * Nin + id]);
    }
    out[((size_t)b * C + c) * Nq + n] = mx;
  }
}

// ---------------------------------------------------------------------------
// Global mean over points: out[b,m] = mean_n in[b,m,n]
// ---------------------------------------------------------------------------
__global__ void mean_kernel(const float* __restrict__ in, int M, int N,
                            float* __restrict__ out) {
  int t = blockIdx.x * 128 + threadIdx.x;
  if (t < 2 * M) {
    int b = t / M, m = t % M;
    const float* p = in + ((size_t)b * M + m) * N;
    float s = 0.f;
    for (int n = 0; n < N; ++n) s += p[n];
    out[t] = s / (float)N;
  }
}

// ---------------------------------------------------------------------------
// Host orchestration
// ---------------------------------------------------------------------------
extern "C" void kernel_launch(void* const* d_in, const int* in_sizes, int n_in,
                              void* d_out, int out_size, void* d_ws, size_t ws_size,
                              hipStream_t stream) {
  (void)out_size; (void)ws_size;
  int cur = 0;
  auto takef = [&]() -> const float* {
    int i = (cur < n_in) ? cur : (n_in - 1);
    ++cur;
    return (const float*)d_in[i];
  };
  const float* x = takef();
  const float* pos = takef();
  int N = in_sizes[1] / 6;  // pos = [2,3,N]
  int N1 = N / 4, N2 = N / 16, N3 = N / 64, N4 = N / 256;

  struct BN { const float *g, *b, *rm, *rv; };
  struct FKAp { const float *fc1, *in1g, *in1b, *fc2, *in2g, *in2b, *fc3, *cvw, *alpha, *beta, *nrad; };
  struct RES { const float *cv0w, *cv0b; BN bn0; FKAp fka; BN bn1;
               const float *cv2w, *cv2b; BN bn2; int has_sc;
               const float *scw, *scb; BN bnsc; };
  auto takeBN = [&]() { BN r; r.g = takef(); r.b = takef(); r.rm = takef(); r.rv = takef(); return r; };
  auto takeFKA = [&]() {
    FKAp r;
    r.fc1 = takef(); r.in1g = takef(); r.in1b = takef();
    r.fc2 = takef(); r.in2g = takef(); r.in2b = takef();
    r.fc3 = takef(); r.cvw = takef();
    r.alpha = takef(); r.beta = takef(); r.nrad = takef();
    return r;
  };
  auto takeRES = [&](int has_sc) {
    RES r;
    r.cv0w = takef(); r.cv0b = takef(); r.bn0 = takeBN();
    r.fka = takeFKA(); r.bn1 = takeBN();
    r.cv2w = takef(); r.cv2b = takef(); r.bn2 = takeBN();
    r.has_sc = has_sc;
    if (has_sc) { r.scw = takef(); r.scb = takef(); r.bnsc = takeBN(); }
    else { r.scw = nullptr; r.scb = nullptr; r.bnsc = BN{nullptr, nullptr, nullptr, nullptr}; }
    return r;
  };
  FKAp cv0 = takeFKA();
  BN bn0p = takeBN();
  RES rb01 = takeRES(0), rb10 = takeRES(1), rb11 = takeRES(0);
  RES rb20 = takeRES(1), rb21 = takeRES(0);
  RES rb30 = takeRES(1), rb31 = takeRES(0);
  RES rb40 = takeRES(1), rb41 = takeRES(0);
  const float* fcoutw = takef();
  const float* fcoutb = takef();

  // -------- workspace bump allocator --------
  char* wsb = (char*)d_ws;
  size_t off = 0;
  auto allocF = [&](size_t n) -> float* {
    float* p = (float*)(wsb + off);
    off += ((n * 4 + 255) / 256) * 256;
    return p;
  };
  auto allocI = [&](size_t n) -> int* {
    int* p = (int*)(wsb + off);
    off += ((n * 4 + 255) / 256) * 256;
    return p;
  };
  float* sup1 = allocF((size_t)2 * 3 * N1);
  float* sup2 = allocF((size_t)2 * 3 * N2);
  float* sup3 = allocF((size_t)2 * 3 * N3);
  float* sup4 = allocF((size_t)2 * 3 * N4);
  int* ids00 = allocI((size_t)2 * N * 16);
  int* ids01 = allocI((size_t)2 * N1 * 16);
  int* ids11 = allocI((size_t)2 * N1 * 16);
  int* ids12 = allocI((size_t)2 * N2 * 16);
  int* ids22 = allocI((size_t)2 * N2 * 16);
  int* ids23 = allocI((size_t)2 * N3 * 16);
  int* ids33 = allocI((size_t)2 * N3 * 16);
  int* ids34 = allocI((size_t)2 * N4 * 16);
  int* ids44 = allocI((size_t)2 * N4 * 16);
  float* mat1 = allocF((size_t)2 * 16 * N * 16);
  float* mat2 = allocF((size_t)2 * 16 * N * 16);
  float* mat3 = allocF((size_t)2 * 16 * N * 16);
  float* dwbuf = allocF((size_t)2 * N * 16);
  float* fsbuf = allocF((size_t)2 * 512 * N);
  float* stats1 = allocF(64);
  float* stats2 = allocF(64);
  float* ss1 = allocF(64);
  float* ss2 = allocF(64);
  float* xa = allocF((size_t)2 * 64 * N);
  float* xb = allocF((size_t)2 * 64 * N);
  float* tbuf = allocF((size_t)2 * 32 * N);
  float* fbuf = allocF((size_t)2 * 32 * N);
  float* xsfull = allocF((size_t)2 * 128 * N);
  float* xspool = allocF((size_t)2 * 128 * N1);
  float* xoutbuf = allocF((size_t)2 * 40 * N4);

  // -------- launch helpers --------
  auto gemm = [&](const float* A, int M, int K, const float* Bmat, int Nn,
                  const float* bias, const BN* bn, const float* res, int relu,
                  float* out) {
    dim3 grid((Nn + 31) / 32, (M + 63) / 64);
    for (int b = 0; b < 2; ++b) {
      gemm_wmma<<<grid, 128, 0, stream>>>(
          A, Bmat + (size_t)b * K * Nn, bias,
          bn ? bn->g : nullptr, bn ? bn->b : nullptr,
          bn ? bn->rm : nullptr, bn ? bn->rv : nullptr,
          res ? res + (size_t)b * M * Nn : nullptr,
          out + (size_t)b * M * Nn, M, Nn, K, relu);
    }
  };

  auto run_fka = [&](const FKAp& P, const float* xf, int C, const float* pts,
                     int Npts, const float* supq, const int* idx, int Nq,
                     const BN* bn, int relu, int M, float* out) {
    (void)hipMemsetAsync(stats1, 0, 64 * sizeof(float), stream);
    (void)hipMemsetAsync(stats2, 0, 64 * sizeof(float), stream);
    dim3 g(Nq / 16, 2);
    fka_stage1<<<g, 256, 0, stream>>>(pts, supq, idx, P.fc1, P.alpha, P.beta,
                                      P.nrad, Npts, Nq, dwbuf, mat1,
                                      stats1, stats1 + 32);
    in_finalize<<<1, 32, 0, stream>>>(stats1, stats1 + 32, P.in1g, P.in1b,
                                      1.f / (float)(Nq * 16), ss1);
    fka_stage_fused<<<g, 256, 0, stream>>>(mat1, dwbuf, ss1, P.fc2, Nq, 0,
                                           mat2, stats2, stats2 + 32);
    in_finalize<<<1, 32, 0, stream>>>(stats2, stats2 + 32, P.in2g, P.in2b,
                                      1.f / (float)(Nq * 16), ss2);
    fka_stage_fused<<<g, 256, 0, stream>>>(mat2, dwbuf, ss2, P.fc3, Nq, 1,
                                           mat3, nullptr, nullptr);
    dim3 gf(Nq, 2);
    fs_kernel<<<gf, 256, 0, stream>>>(xf, C, Npts, idx, mat3, Nq, fsbuf);
    gemm(P.cvw, M, C * 16, fsbuf, Nq, nullptr, bn, nullptr, relu, out);
  };

  auto resblock = [&](const RES& P, int cin, int cout, const float* xin,
                      int NqIn, const float* posIn, const float* supOut,
                      const int* idx, int NqOut, float* out) {
    int ch = cin / 2;
    const float* supq = supOut ? supOut : posIn;
    gemm(P.cv0w, ch, cin, xin, NqIn, P.cv0b, &P.bn0, nullptr, 1, tbuf);
    run_fka(P.fka, tbuf, ch, posIn, NqIn, supq, idx, NqOut, &P.bn1, 1, ch, fbuf);
    const float* xs = xin;
    if (P.has_sc) {
      gemm(P.scw, cout, cin, xin, NqIn, P.scb, &P.bnsc, nullptr, 0, xsfull);
      xs = xsfull;
    }
    if (NqOut != NqIn) {
      int total = 2 * cout * NqOut;
      maxpool_kernel<<<(total + 255) / 256, 256, 0, stream>>>(xs, idx, cout,
                                                              NqIn, NqOut, xspool);
      xs = xspool;
    }
    gemm(P.cv2w, cout, ch, fbuf, NqOut, P.cv2b, &P.bn2, xs, 1, out);
  };

  // -------- network --------
  fps_kernel<<<2, 256, 0, stream>>>(pos, N, N1, sup1);
  fps_kernel<<<2, 256, 0, stream>>>(sup1, N1, N2, sup2);
  fps_kernel<<<2, 256, 0, stream>>>(sup2, N2, N3, sup3);
  fps_kernel<<<2, 256, 0, stream>>>(sup3, N3, N4, sup4);

  auto knn = [&](const float* p, int Np, const float* q, int Nq, int* out) {
    dim3 g((Nq + 255) / 256, 2);
    knn_kernel<<<g, 256, 0, stream>>>(p, Np, q, Nq, out);
  };
  knn(pos, N, pos, N, ids00);
  knn(pos, N, sup1, N1, ids01);
  knn(sup1, N1, sup1, N1, ids11);
  knn(sup1, N1, sup2, N2, ids12);
  knn(sup2, N2, sup2, N2, ids22);
  knn(sup2, N2, sup3, N3, ids23);
  knn(sup3, N3, sup3, N3, ids33);
  knn(sup3, N3, sup4, N4, ids34);
  knn(sup4, N4, sup4, N4, ids44);

  // x0 = relu(bn0(fkaconv(cv0, x, pos, pos, ids00)))
  run_fka(cv0, x, 3, pos, N, pos, ids00, N, &bn0p, 1, 64, xa);
  resblock(rb01, 64, 64, xa, N, pos, nullptr, ids00, N, xb);
  resblock(rb10, 64, 128, xb, N, pos, sup1, ids01, N1, xa);
  resblock(rb11, 128, 128, xa, N1, sup1, nullptr, ids11, N1, xb);
  resblock(rb20, 128, 256, xb, N1, sup1, sup2, ids12, N2, xa);
  resblock(rb21, 256, 256, xa, N2, sup2, nullptr, ids22, N2, xb);
  resblock(rb30, 256, 512, xb, N2, sup2, sup3, ids23, N3, xa);
  resblock(rb31, 512, 512, xa, N3, sup3, nullptr, ids33, N3, xb);
  resblock(rb40, 512, 1024, xb, N3, sup3, sup4, ids34, N4, xa);
  resblock(rb41, 1024, 1024, xa, N4, sup4, nullptr, ids44, N4, xb);

  gemm(fcoutw, 40, 1024, xb, N4, fcoutb, nullptr, nullptr, 0, xoutbuf);
  mean_kernel<<<1, 128, 0, stream>>>(xoutbuf, 40, N4, (float*)d_out);
}
